// TerneryDit_18700287607385
// MI455X (gfx1250) — 8-bit matrix operand order repaired and checked by reading, compiled, not run on hardware
//
#include <hip/hip_runtime.h>
#include <stdint.h>

typedef __attribute__((ext_vector_type(8))) int v8i;

#define EPSF 1e-6f

#define TB_   16
#define SS_   1024
#define TT_   (TB_*SS_)
#define DD_   1024
#define II_   2816
#define LL_   12
#define CC_   768
#define ICC_  2048

#define BM 64
#define BN 128
#define PITCH 80

#define ASYNC_LD16(ldst, gsrc)                                             \
  asm volatile("global_load_async_to_lds_b128 %0, %1, off"                 \
               :: "v"((unsigned)(uintptr_t)(ldst)),                        \
                  "v"((unsigned long long)(uintptr_t)(gsrc))               \
               : "memory")
#define WAIT_ASYNC0() asm volatile("s_wait_asynccnt 0" ::: "memory")

__global__ void k_absmean_partial(const float* __restrict__ w, long long n,
                                  float* __restrict__ partial) {
  __shared__ float red[256];
  float s = 0.f;
  for (long long i = (long long)blockIdx.x*256 + threadIdx.x; i < n;
       i += 256ll * gridDim.x)
    s += fabsf(w[i]);
  red[threadIdx.x] = s; __syncthreads();
  for (int o = 128; o > 0; o >>= 1) {
    if (threadIdx.x < o) red[threadIdx.x] += red[threadIdx.x + o];
    __syncthreads();
  }
  if (threadIdx.x == 0) partial[blockIdx.x] = red[0];
}

__global__ void k_absmean_final(const float* __restrict__ partial, int nb,
                                long long n, float* __restrict__ scale_out) {
  __shared__ float red[256];
  float s = 0.f;
  for (int i = threadIdx.x; i < nb; i += 256) s += partial[i];
  red[threadIdx.x] = s; __syncthreads();
  for (int o = 128; o > 0; o >>= 1) {
    if (threadIdx.x < o) red[threadIdx.x] += red[threadIdx.x + o];
    __syncthreads();
  }
  if (threadIdx.x == 0) *scale_out = fmaxf(red[0] / (float)n, 1e-5f);
}

__global__ void k_quant_w(const float* __restrict__ w,
                          const float* __restrict__ scale, long long n,
                          int8_t* __restrict__ wq) {
  float inv = 1.0f / *scale;
  for (long long i = (long long)blockIdx.x*256 + threadIdx.x; i < n;
       i += 256ll * gridDim.x) {
    float q = rintf(w[i] * inv);
    q = fminf(fmaxf(q, -1.f), 1.f);
    wq[i] = (int8_t)q;
  }
}

__device__ inline float blk_sum(float v, float* red) {
  red[threadIdx.x] = v; __syncthreads();
  for (int o = 128; o > 0; o >>= 1) {
    if (threadIdx.x < o) red[threadIdx.x] += red[threadIdx.x + o];
    __syncthreads();
  }
  float r = red[0]; __syncthreads();
  return r;
}
__device__ inline float blk_max(float v, float* red) {
  red[threadIdx.x] = v; __syncthreads();
  for (int o = 128; o > 0; o >>= 1) {
    if (threadIdx.x < o) red[threadIdx.x] = fmaxf(red[threadIdx.x], red[threadIdx.x + o]);
    __syncthreads();
  }
  float r = red[0]; __syncthreads();
  return r;
}

template <int K>
__global__ __launch_bounds__(256)
void k_act_quant(const float* __restrict__ X,
                 const float* __restrict__ lnw,
                 const float* __restrict__ mod, int tokPerB,
                 int8_t* __restrict__ Xq, float* __restrict__ srecip, int T) {
  constexpr int NP = K / 256;
  __shared__ float red[256];
  int t = blockIdx.x;
  if (t >= T) return;
  const float* x = X + (long long)t * K;
  float r[NP];
  float ss = 0.f;
  #pragma unroll
  for (int j = 0; j < NP; ++j) {
    float v = x[j*256 + threadIdx.x];
    r[j] = v; ss += v * v;
  }
  float inv1 = rsqrtf(blk_sum(ss, red) / (float)K + EPSF);
  int b = t / tokPerB;
  float ss2 = 0.f, mx = 0.f;
  #pragma unroll
  for (int j = 0; j < NP; ++j) {
    int d = j*256 + threadIdx.x;
    float v = r[j];
    if (lnw) v = v * inv1 * lnw[d];
    else     v = v * inv1;
    if (mod) v = v * (1.f + mod[b*2048 + d]) + mod[b*2048 + 1024 + d];
    r[j] = v; ss2 += v * v; mx = fmaxf(mx, fabsf(v));
  }
  float inv2 = rsqrtf(blk_sum(ss2, red) / (float)K + EPSF);
  float mxall = blk_max(mx, red);
  float clipmax = fmaxf(mxall * inv2, 1e-5f);
  float qs = 127.f / clipmax;
  if (threadIdx.x == 0) srecip[t] = clipmax / 127.f;
  #pragma unroll
  for (int j = 0; j < NP; ++j) {
    float xn = r[j] * inv2;
    float q = rintf(xn * qs);
    q = fminf(fmaxf(q, -128.f), 127.f);
    Xq[(long long)t*K + j*256 + threadIdx.x] = (int8_t)q;
  }
}

__device__ inline v8i load_a_frag(const int8_t* ldsA, int mt, int lane) {
  int half = lane >> 4;
  const int8_t* rowp = ldsA + (mt + (lane & 15)) * PITCH;
  v8i a;
  #pragma unroll
  for (int v = 0; v < 8; ++v) {
    int k = ((v >> 1) << 4) + ((v & 1) << 2) + half * 8;
    a[v] = *(const int*)(rowp + k);
  }
  return a;
}
__device__ inline v8i load_b_frag(const int8_t* ldsB, int nt, int lane) {
  int half = lane >> 4;
  const int8_t* rowp = ldsB + (nt + (lane & 15)) * PITCH;
  v8i b;
  #pragma unroll
  for (int v = 0; v < 8; ++v) {
    int k = ((v >> 1) << 4) + ((v & 1) << 2) + half * 8;
    b[v] = *(const int*)(rowp + k);
  }
  return b;
}

__global__ __launch_bounds__(256)
void k_gemm_gate(const int8_t* __restrict__ Xq, const float* __restrict__ xs,
                 const int8_t* __restrict__ Wq, const float* __restrict__ wscale,
                 float* __restrict__ H2, int T, int K, int I) {
  __shared__ __align__(16) int8_t As[2][BM * PITCH];
  __shared__ __align__(16) int8_t Bg[2][BN * PITCH];
  __shared__ __align__(16) int8_t Bu[2][BN * PITCH];
  const int m0 = blockIdx.x * BM;
  const int n0 = blockIdx.y * BN;
  const int tid = threadIdx.x, lane = tid & 31, w = tid >> 5;
  const int mt  = (w & 3) * 16;
  const int nt0 = (w >> 2) * 64;
  v8i cg[4] = {{0,0,0,0,0,0,0,0},{0,0,0,0,0,0,0,0},
               {0,0,0,0,0,0,0,0},{0,0,0,0,0,0,0,0}};
  v8i cu[4] = {{0,0,0,0,0,0,0,0},{0,0,0,0,0,0,0,0},
               {0,0,0,0,0,0,0,0},{0,0,0,0,0,0,0,0}};
  const int nK = K / 64;
  const int lr = tid >> 2, lc = (tid & 3) * 16;

  auto issue = [&](int kb, int buf) {
    const int k0 = kb * 64;
    int row = m0 + lr;
    if (row < T)
      ASYNC_LD16(As[buf] + lr * PITCH + lc,
                 Xq + (long long)row * K + k0 + lc);
    else
      *(uint4*)(As[buf] + lr * PITCH + lc) = make_uint4(0u, 0u, 0u, 0u);
    #pragma unroll
    for (int rr = 0; rr < 2; ++rr) {
      int br = lr + rr * 64;
      ASYNC_LD16(Bg[buf] + br * PITCH + lc,
                 Wq + (long long)(n0 + br) * K + k0 + lc);
      ASYNC_LD16(Bu[buf] + br * PITCH + lc,
                 Wq + (long long)(I + n0 + br) * K + k0 + lc);
    }
  };

  issue(0, 0);
  for (int kb = 0; kb < nK; ++kb) {
    WAIT_ASYNC0();
    __syncthreads();
    const int cur = kb & 1;
    if (kb + 1 < nK) issue(kb + 1, cur ^ 1);
    v8i a   = load_a_frag(As[cur], mt, lane);
    v8i bg0 = load_b_frag(Bg[cur], nt0 +  0, lane);
    v8i bg1 = load_b_frag(Bg[cur], nt0 + 16, lane);
    v8i bg2 = load_b_frag(Bg[cur], nt0 + 32, lane);
    v8i bg3 = load_b_frag(Bg[cur], nt0 + 48, lane);
    v8i bu0 = load_b_frag(Bu[cur], nt0 +  0, lane);
    v8i bu1 = load_b_frag(Bu[cur], nt0 + 16, lane);
    v8i bu2 = load_b_frag(Bu[cur], nt0 + 32, lane);
    v8i bu3 = load_b_frag(Bu[cur], nt0 + 48, lane);
    cg[0] = __builtin_amdgcn_wmma_i32_16x16x64_iu8(true, a, true, bg0, cg[0], false, false);
    cg[1] = __builtin_amdgcn_wmma_i32_16x16x64_iu8(true, a, true, bg1, cg[1], false, false);
    cg[2] = __builtin_amdgcn_wmma_i32_16x16x64_iu8(true, a, true, bg2, cg[2], false, false);
    cg[3] = __builtin_amdgcn_wmma_i32_16x16x64_iu8(true, a, true, bg3, cg[3], false, false);
    cu[0] = __builtin_amdgcn_wmma_i32_16x16x64_iu8(true, a, true, bu0, cu[0], false, false);
    cu[1] = __builtin_amdgcn_wmma_i32_16x16x64_iu8(true, a, true, bu1, cu[1], false, false);
    cu[2] = __builtin_amdgcn_wmma_i32_16x16x64_iu8(true, a, true, bu2, cu[2], false, false);
    cu[3] = __builtin_amdgcn_wmma_i32_16x16x64_iu8(true, a, true, bu3, cu[3], false, false);
  }
  const float wsv = *wscale;
  const int colBase = n0 + nt0 + (lane & 15);
  #pragma unroll
  for (int j = 0; j < 4; ++j) {
    int col = colBase + j * 16;
    #pragma unroll
    for (int r8 = 0; r8 < 8; ++r8) {
      int row = m0 + mt + r8 + ((lane >> 4) << 3);
      if (row < T) {
        float sxw = xs[row] * wsv;
        float g = (float)cg[j][r8] * sxw;
        float u = (float)cu[j][r8] * sxw;
        float sg = g / (1.f + __expf(-g));
        H2[(long long)row * I + col] = sg * u;
      }
    }
  }
}

__global__ __launch_bounds__(256)
void k_gemm_plain(const int8_t* __restrict__ Xq, const float* __restrict__ xs,
                  const int8_t* __restrict__ Wq, const float* __restrict__ wscale,
                  float* __restrict__ Out, const float* __restrict__ Resid,
                  int ldOut, int T, int K) {
  __shared__ __align__(16) int8_t As[2][BM * PITCH];
  __shared__ __align__(16) int8_t Bs[2][BN * PITCH];
  const int m0 = blockIdx.x * BM;
  const int n0 = blockIdx.y * BN;
  const int tid = threadIdx.x, lane = tid & 31, w = tid >> 5;
  const int mt  = (w & 3) * 16;
  const int nt0 = (w >> 2) * 64;
  v8i c[4] = {{0,0,0,0,0,0,0,0},{0,0,0,0,0,0,0,0},
              {0,0,0,0,0,0,0,0},{0,0,0,0,0,0,0,0}};
  const int nK = K / 64;
  const int lr = tid >> 2, lc = (tid & 3) * 16;

  auto issue = [&](int kb, int buf) {
    const int k0 = kb * 64;
    int row = m0 + lr;
    if (row < T)
      ASYNC_LD16(As[buf] + lr * PITCH + lc,
                 Xq + (long long)row * K + k0 + lc);
    else
      *(uint4*)(As[buf] + lr * PITCH + lc) = make_uint4(0u, 0u, 0u, 0u);
    #pragma unroll
    for (int rr = 0; rr < 2; ++rr) {
      int br = lr + rr * 64;
      ASYNC_LD16(Bs[buf] + br * PITCH + lc,
                 Wq + (long long)(n0 + br) * K + k0 + lc);
    }
  };

  issue(0, 0);
  for (int kb = 0; kb < nK; ++kb) {
    WAIT_ASYNC0();
    __syncthreads();
    const int cur = kb & 1;
    if (kb + 1 < nK) issue(kb + 1, cur ^ 1);
    v8i a  = load_a_frag(As[cur], mt, lane);
    v8i b0 = load_b_frag(Bs[cur], nt0 +  0, lane);
    v8i b1 = load_b_frag(Bs[cur], nt0 + 16, lane);
    v8i b2 = load_b_frag(Bs[cur], nt0 + 32, lane);
    v8i b3 = load_b_frag(Bs[cur], nt0 + 48, lane);
    c[0] = __builtin_amdgcn_wmma_i32_16x16x64_iu8(true, a, true, b0, c[0], false, false);
    c[1] = __builtin_amdgcn_wmma_i32_16x16x64_iu8(true, a, true, b1, c[1], false, false);
    c[2] = __builtin_amdgcn_wmma_i32_16x16x64_iu8(true, a, true, b2, c[2], false, false);
    c[3] = __builtin_amdgcn_wmma_i32_16x16x64_iu8(true, a, true, b3, c[3], false, false);
  }
  const float wsv = *wscale;
  const int colBase = n0 + nt0 + (lane & 15);
  #pragma unroll
  for (int j = 0; j < 4; ++j) {
    int col = colBase + j * 16;
    #pragma unroll
    for (int r8 = 0; r8 < 8; ++r8) {
      int row = m0 + mt + r8 + ((lane >> 4) << 3);
      if (row < T) {
        float val = (float)c[j][r8] * xs[row] * wsv;
        if (Resid) val += Resid[(long long)row * ldOut + col];
        Out[(long long)row * ldOut + col] = val;
      }
    }
  }
}

__global__ __launch_bounds__(256)
void k_final_norm(const float* __restrict__ h, const float* __restrict__ fw,
                  float* __restrict__ out) {
  __shared__ float red[256];
  int t = blockIdx.x;
  const float* x = h + (long long)t * DD_;
  float r[4]; float ss = 0.f;
  #pragma unroll
  for (int j = 0; j < 4; ++j) {
    float v = x[j*256 + threadIdx.x]; r[j] = v; ss += v * v;
  }
  float inv = rsqrtf(blk_sum(ss, red) / (float)DD_ + EPSF);
  #pragma unroll
  for (int j = 0; j < 4; ++j) {
    int d = j*256 + threadIdx.x;
    out[(long long)t * DD_ + d] = r[j] * inv * fw[d];
  }
}

extern "C" void kernel_launch(void* const* d_in, const int* in_sizes, int n_in,
                              void* d_out, int out_size, void* d_ws,
                              size_t ws_size, hipStream_t stream) {
  (void)in_sizes; (void)n_in; (void)out_size; (void)ws_size;
  const float* x    = (const float*)d_in[0];
  const float* cond = (const float*)d_in[1];
  const float* agw  = (const float*)d_in[2];
  const float* adw  = (const float*)d_in[3];
  const float* anw  = (const float*)d_in[4];
  const float* aow  = (const float*)d_in[5];
  const float* lnw  = (const float*)d_in[6];
  const float* gw   = (const float*)d_in[7];
  const float* dw   = (const float*)d_in[8];
  const float* fnw  = (const float*)d_in[9];
  float* out = (float*)d_out;

  char* ws = (char*)d_ws;
  size_t off = 0;
  auto alloc = [&](size_t bytes) -> void* {
    void* p = (void*)(ws + off);
    off += (bytes + 255) & ~(size_t)255;
    return p;
  };
  float*  h    = (float*)alloc((size_t)TT_*DD_*4);
  float*  h2   = (float*)alloc((size_t)TT_*II_*4);
  int8_t* xq   = (int8_t*)alloc((size_t)TT_*II_);
  float*  xs1  = (float*)alloc((size_t)TT_*4);
  float*  xs2  = (float*)alloc((size_t)TT_*4);
  int8_t* wqg  = (int8_t*)alloc((size_t)LL_*2*II_*DD_);
  int8_t* wqd  = (int8_t*)alloc((size_t)LL_*DD_*II_);
  int8_t* wqag = (int8_t*)alloc((size_t)2*ICC_*CC_);
  int8_t* wqad = (int8_t*)alloc((size_t)ICC_*ICC_);
  int8_t* wqao = (int8_t*)alloc((size_t)ICC_*ICC_);
  float*  wsc  = (float*)alloc(32*4);
  float*  part = (float*)alloc(256*4);
  float*  params = (float*)alloc((size_t)TB_*ICC_*4);
  float*  ah     = (float*)alloc((size_t)TB_*ICC_*4);

  auto prep = [&](const float* w, long long n, float* slot, int8_t* q) {
    k_absmean_partial<<<256, 256, 0, stream>>>(w, n, part);
    k_absmean_final<<<1, 256, 0, stream>>>(part, 256, n, slot);
    unsigned qb = (unsigned)((n + 256*8 - 1) / (256*8));
    k_quant_w<<<qb, 256, 0, stream>>>(w, slot, n, q);
  };
  prep(agw, (long long)2*ICC_*CC_, wsc + 0, wqag);
  prep(adw, (long long)ICC_*ICC_,  wsc + 1, wqad);
  prep(aow, (long long)ICC_*ICC_,  wsc + 2, wqao);
  for (int l = 0; l < LL_; ++l)
    prep(gw + (long long)l*2*II_*DD_, (long long)2*II_*DD_, wsc + 3 + l,
         wqg + (long long)l*2*II_*DD_);
  for (int l = 0; l < LL_; ++l)
    prep(dw + (long long)l*DD_*II_, (long long)DD_*II_, wsc + 15 + l,
         wqd + (long long)l*DD_*II_);

  k_act_quant<CC_><<<TB_, 256, 0, stream>>>(cond, nullptr, nullptr, 1,
                                            xq, xs1, TB_);
  k_gemm_gate<<<dim3(1, ICC_/BN), 256, 0, stream>>>(xq, xs1, wqag, wsc + 0,
                                                    h2, TB_, CC_, ICC_);
  k_act_quant<ICC_><<<TB_, 256, 0, stream>>>(h2, nullptr, nullptr, 1,
                                             xq, xs2, TB_);
  k_gemm_plain<<<dim3(1, ICC_/BN), 256, 0, stream>>>(xq, xs2, wqad, wsc + 1,
                                                     ah, nullptr, ICC_, TB_, ICC_);
  k_act_quant<ICC_><<<TB_, 256, 0, stream>>>(ah, anw, nullptr, 1,
                                             xq, xs1, TB_);
  k_gemm_plain<<<dim3(1, ICC_/BN), 256, 0, stream>>>(xq, xs1, wqao, wsc + 2,
                                                     params, nullptr, ICC_, TB_, ICC_);

  hipMemcpyAsync(h, x, (size_t)TT_*DD_*4, hipMemcpyDeviceToDevice, stream);
  for (int l = 0; l < LL_; ++l) {
    k_act_quant<DD_><<<TT_, 256, 0, stream>>>(h, lnw + (long long)l*DD_,
                                              params, SS_, xq, xs1, TT_);
    k_gemm_gate<<<dim3(TT_/BM, II_/BN), 256, 0, stream>>>(
        xq, xs1, wqg + (long long)l*2*II_*DD_, wsc + 3 + l, h2, TT_, DD_, II_);
    k_act_quant<II_><<<TT_, 256, 0, stream>>>(h2, nullptr, nullptr, 1,
                                              xq, xs2, TT_);
    k_gemm_plain<<<dim3(TT_/BM, DD_/BN), 256, 0, stream>>>(
        xq, xs2, wqd + (long long)l*DD_*II_, wsc + 15 + l, h, h, DD_, TT_, II_);
  }
  k_final_norm<<<TT_, 256, 0, stream>>>(h, fnw, out);
}
